// Attn_62783831933265
// MI455X (gfx1250) — matrix operand order repaired and checked by reading, not compiled, not run on hardware
//
#include <hip/hip_runtime.h>
#include <hip/hip_bf16.h>

// ---------------------------------------------------------------------------
// CDNA5 (gfx1250) attention block, bf16 WMMA everywhere.
// Shapes: DIM=1024, HQ=16, HK=4, HD=64, B=4, S=2048, TOTAL=8192, NQKV=1536.
// ---------------------------------------------------------------------------

typedef __bf16 bf16_t;
typedef __attribute__((ext_vector_type(16))) __bf16 v16bf;
typedef __attribute__((ext_vector_type(8)))  float  v8f;

union ABFrag { uint4 u[2]; v16bf v; };

static __device__ __forceinline__ v16bf load_frag(const bf16_t* p0, const bf16_t* p1) {
  ABFrag t;
  t.u[0] = *reinterpret_cast<const uint4*>(p0);
  t.u[1] = *reinterpret_cast<const uint4*>(p1);
  return t.v;
}

// ---------------------------------------------------------------------------
// Kernel 1: pre-LayerNorm over DIM=1024 + cast to bf16. One block per row.
// ---------------------------------------------------------------------------
__global__ __launch_bounds__(256) void ln_cast_kernel(
    const float* __restrict__ x, const float* __restrict__ g,
    const float* __restrict__ bta, bf16_t* __restrict__ out) {
  int row = blockIdx.x;
  const float* xr = x + (size_t)row * 1024;
  int t = threadIdx.x;
  float v[4];
  float s = 0.f, sq = 0.f;
#pragma unroll
  for (int i = 0; i < 4; ++i) {
    v[i] = xr[t + i * 256];
    s += v[i];
    sq += v[i] * v[i];
  }
#pragma unroll
  for (int m = 1; m < 32; m <<= 1) { s += __shfl_xor(s, m); sq += __shfl_xor(sq, m); }
  __shared__ float ss[8], ssq[8];
  int wv = t >> 5, lane = t & 31;
  if (lane == 0) { ss[wv] = s; ssq[wv] = sq; }
  __syncthreads();
  s = ss[lane & 7]; sq = ssq[lane & 7];
#pragma unroll
  for (int m = 1; m < 8; m <<= 1) { s += __shfl_xor(s, m); sq += __shfl_xor(sq, m); }
  float mean = s * (1.f / 1024.f);
  float var  = sq * (1.f / 1024.f) - mean * mean;
  float rstd = rsqrtf(var + 1e-5f);
  bf16_t* orow = out + (size_t)row * 1024;
#pragma unroll
  for (int i = 0; i < 4; ++i) {
    int c = t + i * 256;
    orow[c] = (bf16_t)((v[i] - mean) * rstd * g[c] + bta[c]);
  }
}

// ---------------------------------------------------------------------------
// Kernel 2: transpose-cast weights: w (K x N, f32) -> wt (N x K, bf16).
// Row-major N x K means a WMMA B-fragment is two contiguous 16B loads.
// ---------------------------------------------------------------------------
__global__ __launch_bounds__(256) void cvt_transpose_kernel(
    const float* __restrict__ w, bf16_t* __restrict__ wt, int K, int N) {
  int idx = blockIdx.x * 256 + threadIdx.x;
  int n = idx / K, k = idx - n * K;
  wt[idx] = (bf16_t)w[(size_t)k * N + n];
}

// ---------------------------------------------------------------------------
// Kernel 3/6: C(MxN, f32) = A(MxK, bf16 row-major) * BT(NxK, bf16 row-major)^T
// One wave computes a 64x64 tile: 4 M-subtiles x 4 N-subtiles, 16 v8f
// accumulators. Per K=32 step: 8 A-loads + 8 B-loads feed 16 WMMAs
// (1.0 b128 load per v_wmma). Block = 4 waves covering 64 x 256 of C.
// ---------------------------------------------------------------------------
__global__ __launch_bounds__(128) void gemm_bf16_kernel(
    const bf16_t* __restrict__ A, const bf16_t* __restrict__ BT,
    float* __restrict__ C, int M, int N, int K) {
  int lane = threadIdx.x & 31, wv = threadIdx.x >> 5;
  int ml = lane & 15, hi = lane >> 4;
  int m0 = blockIdx.x * 64;
  int n0 = blockIdx.y * 256 + wv * 64;
  const bf16_t* Arow[4];
  const bf16_t* Brow[4];
#pragma unroll
  for (int t = 0; t < 4; ++t) {
    Arow[t] = A + (size_t)(m0 + t * 16 + ml) * K;
    Brow[t] = BT + (size_t)(n0 + t * 16 + ml) * K;
  }
  v8f acc[16] = {};
  for (int kk = 0; kk < K; kk += 32) {
    // 16-bit A layout: lane row=ml, elems K = hi*8+e (e<8) and 16+hi*8+(e-8).
    v16bf a[4];
#pragma unroll
    for (int mt = 0; mt < 4; ++mt)
      a[mt] = load_frag(Arow[mt] + kk + hi * 8, Arow[mt] + kk + 16 + hi * 8);
#pragma unroll
    for (int nt = 0; nt < 4; ++nt) {
      // 16-bit B layout: lane col=ml, elems K = hi*8+e (e<8) and 16+hi*8+(e-8).
      const bf16_t* Br = Brow[nt] + kk + hi * 8;
      v16bf bfr = load_frag(Br, Br + 16);
#pragma unroll
      for (int mt = 0; mt < 4; ++mt)
        acc[mt * 4 + nt] = __builtin_amdgcn_wmma_f32_16x16x32_bf16(
            false, a[mt], false, bfr, (short)0, acc[mt * 4 + nt], false, false);
    }
  }
  // C layout: lane col = ml, row = r + 8*hi in VGPR r.
#pragma unroll
  for (int mt = 0; mt < 4; ++mt)
#pragma unroll
    for (int nt = 0; nt < 4; ++nt)
#pragma unroll
      for (int r = 0; r < 8; ++r)
        C[(size_t)(m0 + mt * 16 + r + 8 * hi) * N + n0 + nt * 16 + ml] =
            acc[mt * 4 + nt][r];
}

// ---------------------------------------------------------------------------
// Kernel 4: per-head LN(Q,K) + RoPE; rearrange to attention layouts.
// One wave handles one (token, head) job: 64 elems = 2 per lane (a RoPE pair).
// jobs per token: 16 q-heads + 4 k-heads (k job also moves its v head).
// Q -> [B,HQ,S,64] ; K -> [B,HK,S,64] (== BT for QK^T) ; V^T -> [B,HK,64,S].
// ---------------------------------------------------------------------------
__global__ __launch_bounds__(128) void qk_rope_kernel(
    const float* __restrict__ qkv, const float* __restrict__ freqs,
    const float* __restrict__ qg, const float* __restrict__ qb,
    const float* __restrict__ kg, const float* __restrict__ kb2,
    bf16_t* __restrict__ Q, bf16_t* __restrict__ Ko, bf16_t* __restrict__ VT) {
  int lane = threadIdx.x & 31, wv = threadIdx.x >> 5;
  int job = blockIdx.x * 4 + wv;
  int token = job / 20, sub = job - token * 20;
  int b = token >> 11, sidx = token & 2047;
  bool isq = sub < 16;
  int h = isq ? sub : sub - 16;
  const float* base = qkv + (size_t)token * 1536 + (isq ? h * 64 : 1024 + h * 64);
  float x0 = base[2 * lane], x1 = base[2 * lane + 1];
  float s = x0 + x1, sq = x0 * x0 + x1 * x1;
#pragma unroll
  for (int m = 1; m < 32; m <<= 1) { s += __shfl_xor(s, m); sq += __shfl_xor(sq, m); }
  float mean = s * (1.f / 64.f);
  float var  = sq * (1.f / 64.f) - mean * mean;
  float rstd = rsqrtf(var + 1e-5f);
  const float* g  = isq ? qg : kg;
  const float* bb = isq ? qb : kb2;
  float y0 = (x0 - mean) * rstd * g[2 * lane] + bb[2 * lane];
  float y1 = (x1 - mean) * rstd * g[2 * lane + 1] + bb[2 * lane + 1];
  float c  = freqs[(size_t)token * 64 + 2 * lane];
  float sn = freqs[(size_t)token * 64 + 2 * lane + 1];
  float o0 = y0 * c - y1 * sn;
  float o1 = y0 * sn + y1 * c;
  if (isq) {
    bf16_t* qr = Q + ((size_t)((b * 16 + h) * 2048 + sidx)) * 64;
    qr[2 * lane] = (bf16_t)o0;
    qr[2 * lane + 1] = (bf16_t)o1;
  } else {
    bf16_t* kr = Ko + ((size_t)((b * 4 + h) * 2048 + sidx)) * 64;
    kr[2 * lane] = (bf16_t)o0;
    kr[2 * lane + 1] = (bf16_t)o1;
    const float* vsrc = qkv + (size_t)token * 1536 + 1280 + h * 64;
    bf16_t* vt = VT + ((size_t)(b * 4 + h)) * 64 * 2048;
    vt[(size_t)(2 * lane) * 2048 + sidx]     = (bf16_t)vsrc[2 * lane];
    vt[(size_t)(2 * lane + 1) * 2048 + sidx] = (bf16_t)vsrc[2 * lane + 1];
  }
}

// ---------------------------------------------------------------------------
// Kernel 5: flash attention. One wave per (b, h, 16-query tile); keys in
// blocks of 32: 4 WMMAs for S = Q K^T, online softmax with 16-lane shfl
// row-reductions, P routed through per-wave LDS (C-layout -> A-layout
// transpose), then 4 WMMAs for O += P V.  Output bf16 [TOTAL, DIM].
// ---------------------------------------------------------------------------
__global__ __launch_bounds__(128) void attn_kernel(
    const bf16_t* __restrict__ Q, const bf16_t* __restrict__ Kb,
    const bf16_t* __restrict__ VT, bf16_t* __restrict__ Oa) {
  __shared__ alignas(16) bf16_t pbuf[4][16][40];  // pitch 40 bf16: 16B-aligned, conflict-free
  int lane = threadIdx.x & 31, wv = threadIdx.x >> 5;
  int tile = blockIdx.x * 4 + wv;     // 0 .. 8191
  int qblk = tile & 127;
  int h    = (tile >> 7) & 15;
  int b    = tile >> 11;
  int hk   = h >> 2;                  // GQA: 4 q-heads per kv-head
  int ml = lane & 15, hi = lane >> 4;
  int s0 = qblk * 16;

  const bf16_t* qrow = Q + ((size_t)((b * 16 + h) * 2048 + s0 + ml)) * 64;
  v16bf a0 = load_frag(qrow + hi * 8,      qrow + 16 + hi * 8);   // dims 0..31
  v16bf a1 = load_frag(qrow + 32 + hi * 8, qrow + 48 + hi * 8);   // dims 32..63
  const bf16_t* kbase = Kb + ((size_t)(b * 4 + hk)) * 2048 * 64;
  const bf16_t* vbase = VT + ((size_t)(b * 4 + hk)) * 64 * 2048;

  v8f oacc[4] = {};
  float Mv[8], Lv[8];
#pragma unroll
  for (int r = 0; r < 8; ++r) { Mv[r] = -1e30f; Lv[r] = 0.f; }
  const float scale = 0.125f;  // 1/sqrt(64)

  for (int kb = 0; kb < 2048; kb += 32) {
    v8f st[2];
#pragma unroll
    for (int t = 0; t < 2; ++t) {
      const bf16_t* kr = kbase + (size_t)(kb + t * 16 + ml) * 64;
      v16bf b0 = load_frag(kr + hi * 8,      kr + 16 + hi * 8);
      v16bf b1 = load_frag(kr + 32 + hi * 8, kr + 48 + hi * 8);
      v8f sacc = {};
      sacc = __builtin_amdgcn_wmma_f32_16x16x32_bf16(false, a0, false, b0, (short)0, sacc, false, false);
      sacc = __builtin_amdgcn_wmma_f32_16x16x32_bf16(false, a1, false, b1, (short)0, sacc, false, false);
      st[t] = sacc;
    }
    float fexp[8];
#pragma unroll
    for (int r = 0; r < 8; ++r) {
      float v0 = st[0][r] * scale, v1 = st[1][r] * scale;
      float mx = fmaxf(v0, v1);
#pragma unroll
      for (int msk = 1; msk < 16; msk <<= 1) mx = fmaxf(mx, __shfl_xor(mx, msk));
      float Mnew = fmaxf(Mv[r], mx);
      float f  = __expf(Mv[r] - Mnew);
      float p0 = __expf(v0 - Mnew), p1 = __expf(v1 - Mnew);
      float rs = p0 + p1;
#pragma unroll
      for (int msk = 1; msk < 16; msk <<= 1) rs += __shfl_xor(rs, msk);
      Lv[r] = Lv[r] * f + rs;
      Mv[r] = Mnew;
      fexp[r] = f;
      pbuf[wv][r + 8 * hi][ml]      = (bf16_t)p0;   // C layout -> LDS row-major
      pbuf[wv][r + 8 * hi][16 + ml] = (bf16_t)p1;
    }
#pragma unroll
    for (int nt = 0; nt < 4; ++nt)
#pragma unroll
      for (int r = 0; r < 8; ++r) oacc[nt][r] *= fexp[r];
    // reload P (16x32) in A layout (per-wave region; LDS is in-order per wave)
    const bf16_t* prow = &pbuf[wv][ml][0];
    v16bf pa = load_frag(prow + hi * 8, prow + 16 + hi * 8);
#pragma unroll
    for (int nt = 0; nt < 4; ++nt) {
      const bf16_t* vr = vbase + (size_t)(nt * 16 + ml) * 2048 + kb + hi * 8;
      v16bf vb = load_frag(vr, vr + 16);
      oacc[nt] = __builtin_amdgcn_wmma_f32_16x16x32_bf16(false, pa, false, vb, (short)0, oacc[nt], false, false);
    }
  }
  int trow = b * 2048 + s0;
#pragma unroll
  for (int nt = 0; nt < 4; ++nt)
#pragma unroll
    for (int r = 0; r < 8; ++r) {
      float val = oacc[nt][r] / Lv[r];
      Oa[(size_t)(trow + r + 8 * hi) * 1024 + h * 64 + nt * 16 + ml] = (bf16_t)val;
    }
}

// ---------------------------------------------------------------------------
// Launcher: ws layout (bytes):
//   h_bf16   @  0MB (16MB) | wqkvT @ 16MB (3MB) | woT @ 19MB (2MB)
//   qkv_f32  @ 21MB (48MB) | Q @ 69MB (16MB) | K @ 85MB (4MB)
//   V^T      @ 89MB (4MB)  | attnO @ 93MB (16MB)   -> 109MB total
// ---------------------------------------------------------------------------
extern "C" void kernel_launch(void* const* d_in, const int* in_sizes, int n_in,
                              void* d_out, int out_size, void* d_ws, size_t ws_size,
                              hipStream_t stream) {
  const float* x     = (const float*)d_in[0];
  const float* freqs = (const float*)d_in[1];
  const float* pre_g = (const float*)d_in[2];
  const float* pre_b = (const float*)d_in[3];
  const float* w_qkv = (const float*)d_in[4];
  const float* q_g   = (const float*)d_in[5];
  const float* q_b   = (const float*)d_in[6];
  const float* k_g   = (const float*)d_in[7];
  const float* k_b   = (const float*)d_in[8];
  const float* w_out = (const float*)d_in[9];
  float* out = (float*)d_out;

  char* ws = (char*)d_ws;
  bf16_t* h     = (bf16_t*)(ws);
  bf16_t* wqkvT = (bf16_t*)(ws + (size_t)16 * 1024 * 1024);
  bf16_t* woT   = (bf16_t*)(ws + (size_t)19 * 1024 * 1024);
  float*  qkv   = (float*) (ws + (size_t)21 * 1024 * 1024);
  bf16_t* Qb    = (bf16_t*)(ws + (size_t)69 * 1024 * 1024);
  bf16_t* Kbuf  = (bf16_t*)(ws + (size_t)85 * 1024 * 1024);
  bf16_t* VTb   = (bf16_t*)(ws + (size_t)89 * 1024 * 1024);
  bf16_t* attnO = (bf16_t*)(ws + (size_t)93 * 1024 * 1024);

  // 1) pre-LN + cast
  ln_cast_kernel<<<8192, 256, 0, stream>>>(x, pre_g, pre_b, h);
  // 2) weight transpose-casts
  cvt_transpose_kernel<<<(1536 * 1024) / 256, 256, 0, stream>>>(w_qkv, wqkvT, 1024, 1536);
  cvt_transpose_kernel<<<(1024 * 1024) / 256, 256, 0, stream>>>(w_out, woT, 1024, 1024);
  // 3) QKV GEMM: (8192x1024) x (1024x1536), 64x256 per block
  gemm_bf16_kernel<<<dim3(128, 6), 128, 0, stream>>>(h, wqkvT, qkv, 8192, 1536, 1024);
  // 4) per-head LN + RoPE + layout shuffle
  qk_rope_kernel<<<40960, 128, 0, stream>>>(qkv, freqs, q_g, q_b, k_g, k_b, Qb, Kbuf, VTb);
  // 5) flash attention (8192 wave-tiles, 4 per block)
  attn_kernel<<<2048, 128, 0, stream>>>(Qb, Kbuf, VTb, attnO);
  // 6) output projection: (8192x1024) x (1024x1024) -> f32 out
  gemm_bf16_kernel<<<dim3(128, 4), 128, 0, stream>>>(attnO, woT, out, 8192, 1024, 1024);
}
